// FeedForward_58025008169031
// MI455X (gfx1250) — compile-verified, matrix operand order checked by reading, not run on hardware
//
#include <hip/hip_runtime.h>
#include <stdint.h>

typedef __attribute__((ext_vector_type(16))) _Float16 v16h;
typedef __attribute__((ext_vector_type(8)))  _Float16 v8h;
typedef __attribute__((ext_vector_type(8)))  float    v8f;

// ---------------------------------------------------------------------------
// Fused A-fragment loader: 8 contiguous channels starting at k0 for row m.
// EPI 0: raw f32 (input x).  EPI 1: f16 raw conv out -> BN+ReLU.
// EPI 2: int8 popcount-conv out -> BN+ReLU+exact GELU.
// sb layout: [scale[0..K), bias[0..K)]
// ---------------------------------------------------------------------------
template<int EPI, int K>
__device__ __forceinline__ void load_act8(const void* __restrict__ A, int m, int k0,
                                          const float* __restrict__ sb, float* o)
{
    if constexpr (EPI == 0) {
        const float* p = (const float*)A + (size_t)m * K + k0;
        #pragma unroll
        for (int j = 0; j < 8; ++j) o[j] = p[j];
    } else if constexpr (EPI == 1) {
        const _Float16* p = (const _Float16*)A + (size_t)m * K + k0;
        v8h r = *(const v8h*)p;                       // 16B aligned contiguous load
        #pragma unroll
        for (int j = 0; j < 8; ++j) {
            float v = (float)r[j] * sb[k0 + j] + sb[K + k0 + j];
            o[j] = v > 0.f ? v : 0.f;
        }
    } else {
        const signed char* p = (const signed char*)A + (size_t)m * K + k0;
        #pragma unroll
        for (int j = 0; j < 8; ++j) {
            float v = (float)p[j] * sb[k0 + j] + sb[K + k0 + j];
            v = v > 0.f ? v : 0.f;
            o[j] = 0.5f * v * (1.f + erff(v * 0.70710678118654752f));   // exact GELU
        }
    }
}

// ---------------------------------------------------------------------------
// One-shot weight prepack: binarize (w>0 ? +1 : -1) into f16, laid out in WMMA
// B-fragment order so a lane loads its whole v16h fragment contiguously:
//   Wbp[((nt*KT + kt)*32 + lane)*16 + j]
//   n = nt*16 + (lane&15);  hi = lane>>4
//   k = kt*32 + (j<8 ? hi*8+j : 16 + hi*8 + (j-8))
// ---------------------------------------------------------------------------
__global__ void prepack_wb_kernel(const float* __restrict__ W,
                                  _Float16* __restrict__ Wbp,
                                  int K, int KT, int total)
{
    int t = blockIdx.x * blockDim.x + threadIdx.x;
    if (t >= total) return;
    int j    = t & 15;
    int lane = (t >> 4) & 31;
    int frag = t >> 9;                  // nt*KT + kt
    int kt   = frag % KT;
    int nt   = frag / KT;
    int n    = nt * 16 + (lane & 15);
    int hi   = lane >> 4;
    int k    = kt * 32 + (j < 8 ? hi * 8 + j : 16 + hi * 8 + (j - 8));
    Wbp[t] = (W[(size_t)n * K + k] > 0.f) ? (_Float16)1.f : (_Float16)-1.f;
}

// ---------------------------------------------------------------------------
// Binary-weight 1x1-conv GEMM:  Y[m][n] = sum_k f(A[m][k]) * sign(W[n][k])
// One wave per 16-row strip; loops over 16-col tiles; K unrolled (K/32 WMMAs).
// B fragments come prepacked (one 32B contiguous load per WMMA, no VALU).
// Also accumulates per-output-channel sum / sum^2 (for the following BN).
// ---------------------------------------------------------------------------
template<int K, int EPI>
__global__ void __launch_bounds__(256)
gemm_bin_kernel(const void* __restrict__ A, const _Float16* __restrict__ Wbp,
                const float* __restrict__ sb, _Float16* __restrict__ Y,
                float* __restrict__ stats, int N)
{
    constexpr int KT = K / 32;
    __shared__ float shs[512];                      // [0..N) sum, [256..256+N) sumsq
    const int tid = threadIdx.x;
    for (int i = tid; i < 512; i += 256) shs[i] = 0.f;
    __syncthreads();

    const int wave = tid >> 5;
    const int lane = tid & 31;
    const int col  = lane & 15;
    const int hi   = lane >> 4;
    const int row0 = (blockIdx.x * 8 + wave) * 16;
    const int mrow = row0 + col;                    // this lane's A row

    __builtin_prefetch((const void*)Wbp, 0, 2);     // weight fragments -> L2 (global_prefetch)

    // Hoist + fuse: epilogue (incl. erf) applied once, reused for all N tiles.
    v16h afrag[KT];
    #pragma unroll
    for (int kt = 0; kt < KT; ++kt) {
        const int kb = kt * 32 + hi * 8;
        float f0[8], f1[8];
        load_act8<EPI, K>(A, mrow, kb,      sb, f0);
        load_act8<EPI, K>(A, mrow, kb + 16, sb, f1);
        #pragma unroll
        for (int j = 0; j < 8; ++j) {
            afrag[kt][j]     = (_Float16)f0[j];
            afrag[kt][8 + j] = (_Float16)f1[j];
        }
    }

    const int N16 = N >> 4;
    for (int nt = 0; nt < N16; ++nt) {
        const int nb   = nt * 16;
        const int ncol = nb + col;
        const v16h* bp = (const v16h*)Wbp + (size_t)nt * KT * 32 + lane;
        v8f acc = {};
        #pragma unroll
        for (int kt = 0; kt < KT; ++kt) {
            v16h bfrag = bp[kt * 32];               // one 32B coalesced load
            acc = __builtin_amdgcn_wmma_f32_16x16x32_f16(
                      false, afrag[kt], false, bfrag, (short)0, acc, false, false);
        }
        // store raw conv output (f16) + per-column stats
        float s0 = 0.f, s1 = 0.f;
        #pragma unroll
        for (int r = 0; r < 8; ++r) {
            float v = acc[r];
            s0 += v; s1 += v * v;
            Y[(size_t)(row0 + hi * 8 + r) * N + ncol] = (_Float16)v;
        }
        s0 += __shfl_down(s0, 16, 32);              // combine row-halves (lane L <- L+16)
        s1 += __shfl_down(s1, 16, 32);
        if (lane < 16) {
            atomicAdd(&shs[nb + lane], s0);
            atomicAdd(&shs[256 + nb + lane], s1);
        }
    }
    __syncthreads();
    for (int i = tid; i < N; i += 256) {
        atomicAdd(&stats[i],     shs[i]);
        atomicAdd(&stats[N + i], shs[256 + i]);
    }
}

// ---------------------------------------------------------------------------
// stats -> per-channel (scale, bias):  scale = g*rsqrt(var+eps), bias = b - mean*scale
// ---------------------------------------------------------------------------
__global__ void scalebias_kernel(const float* __restrict__ stats,
                                 const float* __restrict__ gamma,
                                 const float* __restrict__ beta,
                                 float* __restrict__ sb, int N, float invM)
{
    int c = blockIdx.x * blockDim.x + threadIdx.x;
    if (c >= N) return;
    float mean = stats[c] * invM;
    float var  = stats[N + c] * invM - mean * mean;
    float sc   = gamma[c] * rsqrtf(var + 1e-5f);
    sb[c]     = sc;
    sb[N + c] = beta[c] - mean * sc;
}

// ---------------------------------------------------------------------------
// sign(gelu(relu(bn(y2)))) in {0,1}  ==  (bn(y2) > 0).  Pack 256 ch -> 8 words.
// ---------------------------------------------------------------------------
__global__ void pack_act_kernel(const _Float16* __restrict__ y2,
                                const float* __restrict__ sb,
                                uint32_t* __restrict__ pk, int total)
{
    int t = blockIdx.x * blockDim.x + threadIdx.x;
    if (t >= total) return;
    int m = t >> 3, wd = t & 7;
    const _Float16* p = y2 + (size_t)m * 256 + wd * 32;
    uint32_t bits = 0;
    #pragma unroll
    for (int i = 0; i < 32; ++i) {
        int c = wd * 32 + i;
        float v = (float)p[i] * sb[c] + sb[256 + c];
        if (v > 0.f) bits |= (1u << i);
    }
    pk[t] = bits;                                   // pk[m*8+wd]
}

// Pack 3x3 grouped binary weights: 36 bits per out channel, bit (tap*4+ci).
__global__ void pack_w3_kernel(const float* __restrict__ w3,
                               unsigned long long* __restrict__ w3p)
{
    int oc = threadIdx.x;                           // 256 threads
    unsigned long long bits = 0;
    #pragma unroll
    for (int ci = 0; ci < 4; ++ci)
        for (int t = 0; t < 9; ++t)
            if (w3[((size_t)oc * 4 + ci) * 9 + t] > 0.f)
                bits |= 1ull << (4 * t + ci);
    w3p[oc] = bits;
}

// ---------------------------------------------------------------------------
// 64-group 3x3 binary conv via AND+popcount.  act bits in {0,1}, w in {+-1}:
//   y = 2*popc(w & a) - popc(a).   Zero padding == 0-bits automatically.
// Thread = (pixel, group); 4 out channels/thread; int8 output; BN stats in LDS.
// ---------------------------------------------------------------------------
#define C3_PITER 32
__global__ void __launch_bounds__(256)
conv3x3_bin_kernel(const uint32_t* __restrict__ pk,
                   const unsigned long long* __restrict__ w3p,
                   signed char* __restrict__ y3, float* __restrict__ stats)
{
    __shared__ float shs[512];
    const int tid = threadIdx.x;
    for (int i = tid; i < 512; i += 256) shs[i] = 0.f;
    __syncthreads();

    const int g    = tid & 63;
    const int ps   = tid >> 6;                      // 4 pixels in flight per iter
    const int word = g >> 3;
    const int sh   = (g & 7) * 4;
    unsigned long long wb[4];
    #pragma unroll
    for (int o = 0; o < 4; ++o) wb[o] = w3p[g * 4 + o];

    float s0[4] = {0, 0, 0, 0}, s1[4] = {0, 0, 0, 0};
    for (int it = 0; it < C3_PITER; ++it) {
        const int m = blockIdx.x * (4 * C3_PITER) + it * 4 + ps;   // n*65536+h*256+w
        const int h = (m >> 8) & 255;
        const int w = m & 255;
        unsigned long long a36 = 0;
        int t = 0;
        #pragma unroll
        for (int dy = -1; dy <= 1; ++dy) {
            const int hh = h + dy;
            #pragma unroll
            for (int dx = -1; dx <= 1; ++dx, ++t) {
                const int ww = w + dx;
                if ((unsigned)hh < 256u && (unsigned)ww < 256u) {
                    unsigned nib = (pk[(size_t)(m + dy * 256 + dx) * 8 + word] >> sh) & 0xFu;
                    a36 |= (unsigned long long)nib << (4 * t);
                }
            }
        }
        const int pa = __popcll(a36);
        #pragma unroll
        for (int o = 0; o < 4; ++o) {
            int val = 2 * __popcll(wb[o] & a36) - pa;
            y3[(size_t)m * 256 + g * 4 + o] = (signed char)val;
            float fv = (float)val;
            s0[o] += fv; s1[o] += fv * fv;
        }
    }
    #pragma unroll
    for (int o = 0; o < 4; ++o) {
        atomicAdd(&shs[g * 4 + o], s0[o]);
        atomicAdd(&shs[256 + g * 4 + o], s1[o]);
    }
    __syncthreads();
    {
        int i = tid;
        atomicAdd(&stats[i],       shs[i]);
        atomicAdd(&stats[256 + i], shs[256 + i]);
    }
}

// Final BN+ReLU, f32 out (already NHWC row-major).
__global__ void finalize_kernel(const _Float16* __restrict__ y5,
                                const float* __restrict__ sb,
                                float* __restrict__ out, int total)
{
    int t = blockIdx.x * blockDim.x + threadIdx.x;
    if (t >= total) return;
    int c = t & 63;
    float v = (float)y5[t] * sb[c] + sb[64 + c];
    out[t] = v > 0.f ? v : 0.f;
}

// ---------------------------------------------------------------------------
extern "C" void kernel_launch(void* const* d_in, const int* in_sizes, int n_in,
                              void* d_out, int out_size, void* d_ws, size_t ws_size,
                              hipStream_t stream)
{
    const float* x  = (const float*)d_in[0];
    const float* w1 = (const float*)d_in[1];
    const float* g1 = (const float*)d_in[2];
    const float* b1 = (const float*)d_in[3];
    const float* w2 = (const float*)d_in[4];
    const float* g2 = (const float*)d_in[5];
    const float* b2 = (const float*)d_in[6];
    const float* w3 = (const float*)d_in[7];
    const float* g3 = (const float*)d_in[8];
    const float* b3 = (const float*)d_in[9];
    const float* w4 = (const float*)d_in[10];
    const float* g4 = (const float*)d_in[11];
    const float* b4 = (const float*)d_in[12];
    const float* w5 = (const float*)d_in[13];
    const float* g5 = (const float*)d_in[14];
    const float* b5 = (const float*)d_in[15];

    const int M = in_sizes[0] / 64;                 // 4*256*256 = 262144 rows
    const float invM = 1.f / (float)M;

    char* p = (char*)d_ws;
    auto carve = [&](size_t bytes) -> void* {
        void* r = (void*)p;
        p += (bytes + 255) & ~(size_t)255;
        return r;
    };
    _Float16*           y1  = (_Float16*)carve((size_t)M * 128 * 2);
    _Float16*           y2  = (_Float16*)carve((size_t)M * 256 * 2);
    uint32_t*           pk  = (uint32_t*)carve((size_t)M * 8 * 4);
    signed char*        y3  = (signed char*)carve((size_t)M * 256);
    _Float16*           y4  = (_Float16*)carve((size_t)M * 128 * 2);
    _Float16*           y5  = (_Float16*)carve((size_t)M * 64 * 2);
    float*              st  = (float*)carve(5 * 512 * sizeof(float));
    float*              sb  = (float*)carve(5 * 512 * sizeof(float));
    unsigned long long* w3p = (unsigned long long*)carve(256 * 8);
    _Float16*           wb1 = (_Float16*)carve((size_t)128 * 64  * 2);
    _Float16*           wb2 = (_Float16*)carve((size_t)256 * 128 * 2);
    _Float16*           wb4 = (_Float16*)carve((size_t)128 * 256 * 2);
    _Float16*           wb5 = (_Float16*)carve((size_t)64  * 128 * 2);

    float *st1 = st,       *st2 = st + 512, *st3 = st + 1024, *st4 = st + 1536, *st5 = st + 2048;
    float *sb1 = sb,       *sb2 = sb + 512, *sb3 = sb + 1024, *sb4 = sb + 1536, *sb5 = sb + 2048;

    hipMemsetAsync(st, 0, 5 * 512 * sizeof(float), stream);
    pack_w3_kernel<<<1, 256, 0, stream>>>(w3, w3p);
    prepack_wb_kernel<<<(128 * 64)  / 256, 256, 0, stream>>>(w1, wb1, 64,  2, 128 * 64);
    prepack_wb_kernel<<<(256 * 128) / 256, 256, 0, stream>>>(w2, wb2, 128, 4, 256 * 128);
    prepack_wb_kernel<<<(128 * 256) / 256, 256, 0, stream>>>(w4, wb4, 256, 8, 128 * 256);
    prepack_wb_kernel<<<(64 * 128)  / 256, 256, 0, stream>>>(w5, wb5, 128, 4, 64 * 128);

    // conv1: [M,64] f32 -> [M,128] f16 raw
    gemm_bin_kernel<64, 0><<<M / 128, 256, 0, stream>>>(x, wb1, nullptr, y1, st1, 128);
    scalebias_kernel<<<1, 256, 0, stream>>>(st1, g1, b1, sb1, 128, invM);

    // conv2 (fused BN1+ReLU on load): [M,128] -> [M,256] f16 raw
    gemm_bin_kernel<128, 1><<<M / 128, 256, 0, stream>>>(y1, wb2, sb1, y2, st2, 256);
    scalebias_kernel<<<1, 256, 0, stream>>>(st2, g2, b2, sb2, 256, invM);

    // binary activation (== bn2>0) packed to bits, then popcount 3x3 grouped conv
    pack_act_kernel<<<(M * 8) / 256, 256, 0, stream>>>(y2, sb2, pk, M * 8);
    conv3x3_bin_kernel<<<M / 128, 256, 0, stream>>>(pk, w3p, y3, st3);
    scalebias_kernel<<<1, 256, 0, stream>>>(st3, g3, b3, sb3, 256, invM);

    // conv4 (fused BN3+ReLU+GELU on int8 load): [M,256] -> [M,128] f16 raw
    gemm_bin_kernel<256, 2><<<M / 128, 256, 0, stream>>>(y3, wb4, sb3, y4, st4, 128);
    scalebias_kernel<<<1, 256, 0, stream>>>(st4, g4, b4, sb4, 128, invM);

    // conv5 (fused BN4+ReLU on load): [M,128] -> [M,64] f16 raw
    gemm_bin_kernel<128, 1><<<M / 128, 256, 0, stream>>>(y4, wb5, sb4, y5, st5, 64);
    scalebias_kernel<<<1, 256, 0, stream>>>(st5, g5, b5, sb5, 64, invM);

    // BN5+ReLU -> f32 NHWC output
    finalize_kernel<<<(M * 64) / 256, 256, 0, stream>>>(y5, sb5, (float*)d_out, M * 64);
}
